// FasterRCNNResnet50B18_67877663146182
// MI455X (gfx1250) — matrix operand order repaired and checked by reading, not compiled, not run on hardware
//
#include <hip/hip_runtime.h>
#include <stdint.h>

// ---------------- problem constants ----------------
#define HFD 100
#define WFD 100
#define PPH 102                  // padded (halo=1) feature height
#define PPW 102
#define CCH 256
#define NPIX (HFD*WFD)           // 10000
#define PPIX (PPH*PPW)           // 10404
#define BATCH 2
#define AANCH 3
#define NANCH (NPIX*AANCH)       // 30000
#define PRE_NMS 2000
#define POST_NMS 1000
#define NMS_TH 0.7f
#define POOLN 7
#define SRN 2
#define REP 1024
#define FCK (CCH*POOLN*POOLN)    // 12544
#define IMGSZ 800.0f
#define BBOX_CLAMP 4.135166556742356f   // log(1000/16)

#define LDS_STRIDE 264           // 256 k-elems + 8 pad elems (16B) per row

typedef __attribute__((ext_vector_type(16))) __bf16 v16bf;
typedef __attribute__((ext_vector_type(8)))  float  v8f;
typedef __attribute__((ext_vector_type(4)))  uint32_t v4u;
typedef __attribute__((ext_vector_type(8)))  int      v8i;
typedef __attribute__((ext_vector_type(4)))  int      v4i;

union Frag16 { v16bf v; uint4 q[2]; };

__device__ __forceinline__ uint16_t f2bf(float f) {
  uint32_t u = __float_as_uint(f);
  u += 0x7FFFu + ((u >> 16) & 1u);           // round-to-nearest-even
  return (uint16_t)(u >> 16);
}
__device__ __forceinline__ float bf2f(uint16_t h) {
  return __uint_as_float(((uint32_t)h) << 16);
}
__device__ __forceinline__ uint4 ld128(const uint16_t* p) {
  return *reinterpret_cast<const uint4*>(p);
}

// ---------------- conversion kernels ----------------

// NCHW f32 -> zero-padded NHWC bf16 (102x102 with 1-px halo)
__global__ void k_cvt_featpad(const float* __restrict__ in, uint16_t* __restrict__ out) {
  int idx = blockIdx.x * blockDim.x + threadIdx.x;
  int n = BATCH * PPIX * CCH;
  if (idx >= n) return;
  int c  = idx % CCH;
  int px = (idx / CCH) % PPW;
  int py = (idx / (CCH * PPW)) % PPH;
  int b  = idx / (CCH * PPW * PPH);
  int y = py - 1, x = px - 1;
  float v = 0.0f;
  if (y >= 0 && y < HFD && x >= 0 && x < WFD)
    v = in[((size_t)(b * CCH + c)) * NPIX + y * WFD + x];
  out[idx] = f2bf(v);
}

// OIHW (256,256,3,3) f32 -> [tap][O][I] bf16
__global__ void k_cvt_convw(const float* __restrict__ in, uint16_t* __restrict__ out) {
  int idx = blockIdx.x * blockDim.x + threadIdx.x;
  int n = 9 * CCH * CCH;
  if (idx >= n) return;
  int ci  = idx % CCH;
  int o   = (idx / CCH) % CCH;
  int tap = idx / (CCH * CCH);
  out[idx] = f2bf(in[((size_t)(o * CCH + ci)) * 9 + tap]);
}

// (K,N) f32 row-major -> (N,K) bf16 row-major (transpose)
__global__ void k_cvt_fcT(const float* __restrict__ in, uint16_t* __restrict__ out,
                          int K, int N) {
  int idx = blockIdx.x * blockDim.x + threadIdx.x;
  if (idx >= K * N) return;
  int k = idx % K;
  int nn = idx / K;
  out[idx] = f2bf(in[(size_t)k * N + nn]);
}

// ---------------- 3x3 conv (implicit GEMM, WMMA bf16, halo-padded input) ----------------
__global__ void k_conv3_wmma(const uint16_t* __restrict__ featP,
                             const uint16_t* __restrict__ wT,
                             const float* __restrict__ bias,
                             uint16_t* __restrict__ outN) {
  const int lane = threadIdx.x & 31;
  const int wave = threadIdx.x >> 5;
  const int oc0 = (blockIdx.y * 4 + wave) * 16;    // 16 tiles of out channels
  const int m0  = blockIdx.x * 16;                 // pixel tile within image
  const int b   = blockIdx.z;
  const bool hi = lane >= 16;
  const int l15 = lane & 15;

  const int apix = m0 + l15;
  const int ay = apix / WFD, ax = apix % WFD;

  v8f acc = {};
  for (int tap = 0; tap < 9; ++tap) {
    const int dy = tap / 3, dx = tap % 3;          // padded coords: (ay+dy, ax+dx)
    const uint16_t* abase = featP +
        ((size_t)(b * PPIX + (ay + dy) * PPW + (ax + dx))) * CCH + (hi ? 8 : 0);
    const uint16_t* bbase = wT + ((size_t)tap * CCH + oc0 + l15) * CCH + (hi ? 8 : 0);
    #pragma unroll
    for (int kc = 0; kc < CCH; kc += 32) {
      Frag16 fa, fb;
      fa.q[0] = ld128(abase + kc);
      fa.q[1] = ld128(abase + kc + 16);
      fb.q[0] = ld128(bbase + kc);
      fb.q[1] = ld128(bbase + kc + 16);
      acc = __builtin_amdgcn_wmma_f32_16x16x32_bf16(
          false, fa.v, false, fb.v, (short)0, acc, false, false);
    }
  }
  const int oc = oc0 + l15;
  const float bv = bias[oc];
  #pragma unroll
  for (int r = 0; r < 8; ++r) {
    const int prow = m0 + r + (hi ? 8 : 0);
    float v = fmaxf(acc[r] + bv, 0.0f);
    outN[((size_t)(b * NPIX + prow)) * CCH + oc] = f2bf(v);
  }
}

// ---------------- RPN 1x1 heads ----------------
__global__ void k_rpn_heads(const uint16_t* __restrict__ tN,
                            const float* __restrict__ clsW, const float* __restrict__ clsB,
                            const float* __restrict__ boxW, const float* __restrict__ boxB,
                            float* __restrict__ obj, float* __restrict__ dl) {
  int idx = blockIdx.x * blockDim.x + threadIdx.x;
  int n = BATCH * NPIX * 15;
  if (idx >= n) return;
  int o15 = idx % 15;
  int p = (idx / 15) % NPIX;
  int b = idx / (15 * NPIX);
  const uint16_t* trow = tN + ((size_t)(b * NPIX + p)) * CCH;
  const float* wrow;
  float bv;
  if (o15 < 3) { wrow = clsW + o15 * CCH; bv = clsB[o15]; }
  else         { int j = o15 - 3; wrow = boxW + j * CCH; bv = boxB[j]; }
  float s = 0.0f;
  for (int k = 0; k < CCH; ++k) s += bf2f(trow[k]) * wrow[k];
  s += bv;
  if (o15 < 3) {
    obj[(size_t)b * NANCH + p * 3 + o15] = s;
  } else {
    int j = o15 - 3;
    dl[((size_t)b * NANCH + p * 3 + (j >> 2)) * 4 + (j & 3)] = s;
  }
}

// ---------------- exact top-k via stable rank counting ----------------
__global__ void k_topk(const float* __restrict__ obj,
                       int* __restrict__ tki, float* __restrict__ tks) {
  int idx = blockIdx.x * blockDim.x + threadIdx.x;
  if (idx >= BATCH * NANCH) return;
  int i = idx % NANCH;
  int b = idx / NANCH;
  const float* s = obj + (size_t)b * NANCH;
  float si = s[i];
  int cnt = 0;
  for (int j = 0; j < NANCH; ++j) {
    float sj = s[j];
    cnt += (sj > si) || (sj == si && j < i);
  }
  if (cnt < PRE_NMS) {
    tki[b * PRE_NMS + cnt] = i;
    tks[b * PRE_NMS + cnt] = si;
  }
}

// ---------------- anchor decode + clip ----------------
__global__ void k_decode(const int* __restrict__ tki, const float* __restrict__ dl,
                         float* __restrict__ boxes) {
  int idx = blockIdx.x * blockDim.x + threadIdx.x;
  if (idx >= BATCH * PRE_NMS) return;
  int p = idx % PRE_NMS;
  int b = idx / PRE_NMS;
  int i = tki[b * PRE_NMS + p];
  int pix = i / 3, a = i % 3;
  int y = pix / WFD, x = pix % WFD;
  float ra = (a == 0) ? 0.5f : ((a == 1) ? 1.0f : 2.0f);
  float sq = sqrtf(ra);
  float ha = 64.0f * sq;
  float wa = 64.0f / sq;
  float cxa = (float)x * 8.0f;
  float cya = (float)y * 8.0f;
  const float* d = dl + ((size_t)b * NANCH + i) * 4;
  float dw = fminf(d[2], BBOX_CLAMP);
  float dh = fminf(d[3], BBOX_CLAMP);
  float cx = d[0] * wa + cxa;
  float cy = d[1] * ha + cya;
  float w = expf(dw) * wa;
  float h = expf(dh) * ha;
  float* o = boxes + ((size_t)b * PRE_NMS + p) * 4;
  o[0] = fminf(fmaxf(cx - 0.5f * w, 0.0f), IMGSZ);
  o[1] = fminf(fmaxf(cy - 0.5f * h, 0.0f), IMGSZ);
  o[2] = fminf(fmaxf(cx + 0.5f * w, 0.0f), IMGSZ);
  o[3] = fminf(fmaxf(cy + 0.5f * h, 0.0f), IMGSZ);
}

// ---------------- sequential NMS (boxes already score-sorted) ----------------
__global__ void k_nms(const float* __restrict__ boxes, float* __restrict__ rois) {
  __shared__ float bx[PRE_NMS * 4];
  __shared__ unsigned char alive[PRE_NMS];
  const int b = blockIdx.x;
  const int tid = threadIdx.x;
  const int nt = blockDim.x;
  for (int i = tid; i < PRE_NMS * 4; i += nt) bx[i] = boxes[(size_t)b * PRE_NMS * 4 + i];
  for (int i = tid; i < PRE_NMS; i += nt) alive[i] = 1;
  __syncthreads();
  for (int i = 0; i < PRE_NMS; ++i) {
    if (alive[i]) {
      float xi1 = bx[4*i], yi1 = bx[4*i+1], xi2 = bx[4*i+2], yi2 = bx[4*i+3];
      float ai = (xi2 - xi1) * (yi2 - yi1);
      for (int j = i + 1 + tid; j < PRE_NMS; j += nt) {
        if (!alive[j]) continue;
        float xj1 = bx[4*j], yj1 = bx[4*j+1], xj2 = bx[4*j+2], yj2 = bx[4*j+3];
        float lx = fmaxf(xi1, xj1), ly = fmaxf(yi1, yj1);
        float rx = fminf(xi2, xj2), ry = fminf(yi2, yj2);
        float iw = fmaxf(rx - lx, 0.0f), ih = fmaxf(ry - ly, 0.0f);
        float inter = iw * ih;
        float aj = (xj2 - xj1) * (yj2 - yj1);
        float iou = inter / (ai + aj - inter + 1e-9f);
        if (iou > NMS_TH) alive[j] = 0;
      }
    }
    __syncthreads();
  }
  if (tid == 0) {
    int k = 0;
    for (int j = 0; j < PRE_NMS && k < POST_NMS; ++j) {
      if (alive[j]) {
        float* o = rois + ((size_t)b * POST_NMS + k) * 4;
        o[0] = bx[4*j]; o[1] = bx[4*j+1]; o[2] = bx[4*j+2]; o[3] = bx[4*j+3];
        ++k;
      }
    }
    for (int j = 0; j < PRE_NMS && k < POST_NMS; ++j) {
      if (!alive[j]) {
        float* o = rois + ((size_t)b * POST_NMS + k) * 4;
        o[0] = bx[4*j]; o[1] = bx[4*j+1]; o[2] = bx[4*j+2]; o[3] = bx[4*j+3];
        ++k;
      }
    }
  }
}

// ---------------- RoI align on padded features; GEMM-ready bf16 output ----------------
__global__ void k_roialign(const uint16_t* __restrict__ featP,
                           const float* __restrict__ rois,
                           uint16_t* __restrict__ pooled) {
  __shared__ float sxs[POOLN * SRN], sys[POOLN * SRN];
  const int roi = blockIdx.x;
  const int b = blockIdx.y;
  const float* r = rois + ((size_t)b * POST_NMS + roi) * 4;
  const float x1 = r[0] / 8.0f, y1 = r[1] / 8.0f, x2 = r[2] / 8.0f, y2 = r[3] / 8.0f;
  const float bw = (x2 - x1) / (float)POOLN;
  const float bh = (y2 - y1) / (float)POOLN;
  if (threadIdx.x < POOLN * SRN) {
    float k = ((float)threadIdx.x + 0.5f) / (float)SRN;
    sxs[threadIdx.x] = fminf(fmaxf(x1 + bw * k, 0.0f), (float)(WFD - 1));
    sys[threadIdx.x] = fminf(fmaxf(y1 + bh * k, 0.0f), (float)(HFD - 1));
  }
  __syncthreads();
  const int c = threadIdx.x;   // 256 channels
  const uint16_t* fb = featP + (size_t)b * PPIX * CCH;
  uint16_t* prow = pooled + ((size_t)(b * POST_NMS + roi)) * FCK + c * (POOLN * POOLN);
  for (int ph = 0; ph < POOLN; ++ph) {
    for (int pw = 0; pw < POOLN; ++pw) {
      float acc = 0.0f;
      #pragma unroll
      for (int sy = 0; sy < SRN; ++sy) {
        float yv = sys[ph * SRN + sy];
        int y0i = (int)floorf(yv);
        int y1i = min(y0i + 1, HFD - 1);
        float ly = yv - floorf(yv), hy = 1.0f - ly;
        #pragma unroll
        for (int sx = 0; sx < SRN; ++sx) {
          float xv = sxs[pw * SRN + sx];
          int x0i = (int)floorf(xv);
          int x1i = min(x0i + 1, WFD - 1);
          float lx = xv - floorf(xv), hx = 1.0f - lx;
          float v00 = bf2f(fb[((size_t)((y0i+1) * PPW + (x0i+1))) * CCH + c]);
          float v01 = bf2f(fb[((size_t)((y0i+1) * PPW + (x1i+1))) * CCH + c]);
          float v10 = bf2f(fb[((size_t)((y1i+1) * PPW + (x0i+1))) * CCH + c]);
          float v11 = bf2f(fb[((size_t)((y1i+1) * PPW + (x1i+1))) * CCH + c]);
          acc += hy * hx * v00 + hy * lx * v01 + ly * hx * v10 + ly * lx * v11;
        }
      }
      prow[ph * POOLN + pw] = f2bf(acc * 0.25f);
    }
  }
}

// ---------------- bf16 WMMA GEMM with TDM-staged A panels ----------------
// A: MxK bf16 row-major. Wt: NxK bf16 (W transposed). D = act(A@W + bias): MxN bf16.
// Block = 4 waves sharing one 16-row A panel (16 x 256 k-elems) staged into LDS by the
// Tensor Data Mover (double-buffered); each wave owns one 16-col N tile.
// Requires K % 256 == 0, M % 16 == 0.
__global__ void k_gemm_wmma(const uint16_t* __restrict__ A,
                            const uint16_t* __restrict__ Wt,
                            const float* __restrict__ bias,
                            uint16_t* __restrict__ D,
                            int M, int N, int K, int relu) {
  __shared__ __align__(16) uint16_t sA[2][16 * LDS_STRIDE];
  const int tid  = threadIdx.x;
  const int lane = tid & 31;
  const int wave = tid >> 5;
  const int n0 = (blockIdx.y * 4 + wave) * 16;
  const int m0 = blockIdx.x * 16;
  const bool hi = lane >= 16;
  const int l15 = lane & 15;

  // TDM descriptor issue: load a 16 x 256 bf16 tile (row stride K elems) into LDS
  // with native padding (pad_interval = 512B row, pad_amount = 16B) -> LDS_STRIDE rows.
  auto tdm_issue = [&](int kp, int buf) {
    uint64_t ga = (uint64_t)(uintptr_t)A + (((uint64_t)m0 * (uint64_t)K + (uint64_t)kp) << 1);
    uint32_t lds = (uint32_t)(uintptr_t)&sA[buf][0];
    v4u g0 = { 1u,                                   // count=1 (valid user descriptor)
               lds,                                  // lds_addr
               (uint32_t)ga,                         // global_addr[31:0]
               (uint32_t)((ga >> 32) & 0x1FFFFFFu) | (2u << 30) };  // addr[56:32] | type=2
    const uint32_t td0 = 0x40000000u;                // tensor_dim0 (huge: no OOB clip)
    const uint32_t td1 = 0x00100000u;                // tensor_dim1 (huge)
    uint64_t st0 = (uint64_t)(uint32_t)K;            // tensor_dim0_stride (elems)
    v8i g1 = {
      (int)((1u << 16) | (1u << 20) | (6u << 22) | (3u << 25)), // data=2B,pad_en,ivl=512B,amt=16B
      (int)((td0 & 0xFFFFu) << 16),                             // [15:0] bar addr=0, [31:16] td0.lo
      (int)((td0 >> 16) | ((td1 & 0xFFFFu) << 16)),             // td0.hi | td1.lo
      (int)((td1 >> 16) | (256u << 16)),                        // td1.hi | tile_dim0=256
      (int)(16u),                                               // tile_dim1=16, tile_dim2=0
      (int)(uint32_t)st0,                                       // stride0[31:0]
      (int)(uint32_t)(st0 >> 32),                               // stride0[47:32] | stride1.lo=0
      0 };
    v4i z4 = { 0, 0, 0, 0 };
    v8i z8 = { 0, 0, 0, 0, 0, 0, 0, 0 };
    __builtin_amdgcn_tensor_load_to_lds(g0, g1, z4, z4, z8, 0);
  };

  const uint16_t* brow = Wt + (size_t)(n0 + l15) * K + (hi ? 8 : 0);
  const int nPanels = K >> 8;

  if (tid < 32) {
    tdm_issue(0, 0);
    __builtin_amdgcn_s_wait_tensorcnt(0);
  }
  __syncthreads();

  v8f acc = {};
  for (int p = 0; p < nPanels; ++p) {
    const int buf = p & 1;
    if (tid < 32 && (p + 1) < nPanels) tdm_issue((p + 1) << 8, buf ^ 1);

    const uint16_t* ap0 = &sA[buf][l15 * LDS_STRIDE + (hi ? 8 : 0)];
    const uint16_t* bp0 = brow + (p << 8);
    __builtin_prefetch(bp0 + 256, 0, 1);             // stream-prefetch next weight panel
    #pragma unroll
    for (int ks = 0; ks < 8; ++ks) {
      Frag16 fa, fb;
      fa.q[0] = ld128(ap0 + ks * 32);                // ds_load_b128 from staged panel
      fa.q[1] = ld128(ap0 + ks * 32 + 16);
      fb.q[0] = ld128(bp0 + ks * 32);
      fb.q[1] = ld128(bp0 + ks * 32 + 16);
      acc = __builtin_amdgcn_wmma_f32_16x16x32_bf16(
          false, fa.v, false, fb.v, (short)0, acc, false, false);
    }
    if (tid < 32) __builtin_amdgcn_s_wait_tensorcnt(0);
    __syncthreads();
  }

  const int col = n0 + l15;
  const float bv = bias[col];
  #pragma unroll
  for (int r = 0; r < 8; ++r) {
    const int row = m0 + r + (hi ? 8 : 0);
    float v = acc[r] + bv;
    if (relu) v = fmaxf(v, 0.0f);
    D[(size_t)row * N + col] = f2bf(v);
  }
}

// ---------------- final cls/reg heads ----------------
__global__ void k_final(const uint16_t* __restrict__ x2,
                        const float* __restrict__ clsW, const float* __restrict__ clsB,
                        const float* __restrict__ regW, const float* __restrict__ regB,
                        float* __restrict__ out) {
  int idx = blockIdx.x * blockDim.x + threadIdx.x;
  int n = BATCH * POST_NMS * 10;
  if (idx >= n) return;
  int o = idx % 10;
  int row = idx / 10;
  const uint16_t* xr = x2 + (size_t)row * REP;
  float s;
  if (o < 2) {
    s = clsB[o];
    for (int k = 0; k < REP; ++k) s += bf2f(xr[k]) * clsW[k * 2 + o];
  } else {
    int j = o - 2;
    s = regB[j];
    for (int k = 0; k < REP; ++k) s += bf2f(xr[k]) * regW[k * 8 + j];
  }
  out[(size_t)row * 10 + o] = s;
}

// ---------------- launcher ----------------
extern "C" void kernel_launch(void* const* d_in, const int* in_sizes, int n_in,
                              void* d_out, int out_size, void* d_ws, size_t ws_size,
                              hipStream_t stream) {
  const float* features   = (const float*)d_in[0];
  const float* rpn_conv_w = (const float*)d_in[1];
  const float* rpn_conv_b = (const float*)d_in[2];
  const float* rpn_cls_w  = (const float*)d_in[3];
  const float* rpn_cls_b  = (const float*)d_in[4];
  const float* rpn_bbox_w = (const float*)d_in[5];
  const float* rpn_bbox_b = (const float*)d_in[6];
  const float* fc1_w      = (const float*)d_in[7];
  const float* fc1_b      = (const float*)d_in[8];
  const float* fc2_w      = (const float*)d_in[9];
  const float* fc2_b      = (const float*)d_in[10];
  const float* cls_w      = (const float*)d_in[11];
  const float* cls_b      = (const float*)d_in[12];
  const float* reg_w      = (const float*)d_in[13];
  const float* reg_b      = (const float*)d_in[14];

  char* ws = (char*)d_ws;
  size_t off = 0;
  auto carve = [&](size_t bytes) -> char* {
    char* p = ws + off;
    off = (off + bytes + 255) & ~(size_t)255;
    return p;
  };
  uint16_t* featP  = (uint16_t*)carve((size_t)BATCH * PPIX * CCH * 2);
  uint16_t* tN     = (uint16_t*)carve((size_t)BATCH * NPIX * CCH * 2);
  uint16_t* convwT = (uint16_t*)carve((size_t)9 * CCH * CCH * 2);
  uint16_t* fc1T   = (uint16_t*)carve((size_t)REP * FCK * 2);
  uint16_t* fc2T   = (uint16_t*)carve((size_t)REP * REP * 2);
  float*    obj    = (float*)carve((size_t)BATCH * NANCH * 4);
  float*    dl     = (float*)carve((size_t)BATCH * NANCH * 4 * 4);
  int*      tki    = (int*)carve((size_t)BATCH * PRE_NMS * 4);
  float*    tks    = (float*)carve((size_t)BATCH * PRE_NMS * 4);
  float*    boxes  = (float*)carve((size_t)BATCH * PRE_NMS * 4 * 4);
  float*    rois   = (float*)carve((size_t)BATCH * POST_NMS * 4 * 4);
  uint16_t* pooled = (uint16_t*)carve((size_t)BATCH * POST_NMS * FCK * 2);
  uint16_t* x1     = (uint16_t*)carve((size_t)BATCH * POST_NMS * REP * 2);
  uint16_t* x2     = (uint16_t*)carve((size_t)BATCH * POST_NMS * REP * 2);

  const int TB = 256;
  {
    int n = BATCH * PPIX * CCH;
    k_cvt_featpad<<<(n + TB - 1) / TB, TB, 0, stream>>>(features, featP);
  }
  {
    int n = 9 * CCH * CCH;
    k_cvt_convw<<<(n + TB - 1) / TB, TB, 0, stream>>>(rpn_conv_w, convwT);
  }
  {
    int n = FCK * REP;
    k_cvt_fcT<<<(n + TB - 1) / TB, TB, 0, stream>>>(fc1_w, fc1T, FCK, REP);
  }
  {
    int n = REP * REP;
    k_cvt_fcT<<<(n + TB - 1) / TB, TB, 0, stream>>>(fc2_w, fc2T, REP, REP);
  }
  k_conv3_wmma<<<dim3(NPIX / 16, 4, BATCH), 128, 0, stream>>>(featP, convwT, rpn_conv_b, tN);
  {
    int n = BATCH * NPIX * 15;
    k_rpn_heads<<<(n + TB - 1) / TB, TB, 0, stream>>>(tN, rpn_cls_w, rpn_cls_b,
                                                      rpn_bbox_w, rpn_bbox_b, obj, dl);
  }
  {
    int n = BATCH * NANCH;
    k_topk<<<(n + TB - 1) / TB, TB, 0, stream>>>(obj, tki, tks);
  }
  {
    int n = BATCH * PRE_NMS;
    k_decode<<<(n + TB - 1) / TB, TB, 0, stream>>>(tki, dl, boxes);
  }
  k_nms<<<BATCH, 1024, 0, stream>>>(boxes, rois);
  k_roialign<<<dim3(POST_NMS, BATCH), 256, 0, stream>>>(featP, rois, pooled);
  k_gemm_wmma<<<dim3((BATCH * POST_NMS) / 16, REP / 64), 128, 0, stream>>>(
      pooled, fc1T, fc1_b, x1, BATCH * POST_NMS, REP, FCK, 1);
  k_gemm_wmma<<<dim3((BATCH * POST_NMS) / 16, REP / 64), 128, 0, stream>>>(
      x1, fc2T, fc2_b, x2, BATCH * POST_NMS, REP, REP, 1);
  {
    int n = BATCH * POST_NMS * 10;
    k_final<<<(n + TB - 1) / TB, TB, 0, stream>>>(x2, cls_w, cls_b, reg_w, reg_b,
                                                  (float*)d_out);
  }
  (void)in_sizes; (void)n_in; (void)out_size; (void)ws_size; (void)tks;
}
